// GATModel_81595788689804
// MI455X (gfx1250) — matrix operand order repaired and checked by reading, not compiled, not run on hardware
//
#include <hip/hip_runtime.h>
#include <hip/hip_bf16.h>
#include <math.h>

#define F_IN   128
#define F_OUT  64
#define RNUM   11
#define ALPHA  0.2f
#define ZSCALE 4096.0f
#define INV_ZSCALE (1.0f/4096.0f)
#define NBLK   1024            // fixed grid for reduction kernels (determinism)

typedef __attribute__((ext_vector_type(4)))  _Float16 h4;
typedef __attribute__((ext_vector_type(8)))  _Float16 h8;
typedef __attribute__((ext_vector_type(16))) _Float16 h16;
typedef __attribute__((ext_vector_type(8)))  float    v8f;

// ---- workspace layout (byte offsets) ----
#define META_OFF 0          // 2048 B of int32 metadata (memset to 0 each call)
#define UV_OFF   4096       // u[11][128], v[11][128] f32
#define WP_OFF   20480      // packed f16 W fragments: 11*4*4*32*16 halves = 180224 B
#define PS_OFF   262144     // partial col sums: NBLK*128 f32 = 512 KB
#define SORT_OFF 786432     // sorted edge ids: E int32
#define ZH_OFF   2097152    // z as f16: E*128 halves = 51.2 MB

// meta int indices
#define MI_CNT  0
#define MI_CUR  16
#define MI_BASE 32
#define MI_TP   48          // tile prefix, tp[RNUM] = total tiles
#define MI_CMAX 64          // unsigned[128] encoded column max
#define MI_CSUM 192         // float[128] column sum

__device__ __forceinline__ unsigned enc_f32(float f) {
    int i = __float_as_int(f);
    return (i >= 0) ? ((unsigned)i | 0x80000000u) : ~(unsigned)i;
}
__device__ __forceinline__ float dec_f32(unsigned u) {
    int i = (u & 0x80000000u) ? (int)(u & 0x7FFFFFFFu) : ~(int)u;
    return __int_as_float(i);
}
__device__ __forceinline__ float lrelu(float s) { return s > 0.0f ? s : ALPHA * s; }

// K0: u[r][a] = sum_b W[r][a][b]*a1[b], v likewise with a2.  <<<RNUM,128>>>
__global__ void k_uv(const float* __restrict__ W, const float* __restrict__ aw,
                     float* __restrict__ uv) {
    int r = blockIdx.x, a = threadIdx.x;
    const float* wr = W + ((size_t)r * F_IN + a) * F_OUT;
    float s1 = 0.f, s2 = 0.f;
    for (int b = 0; b < F_OUT; ++b) { s1 += wr[b] * aw[b]; s2 += wr[b] * aw[F_OUT + b]; }
    uv[r * F_IN + a] = s1;
    uv[RNUM * F_IN + r * F_IN + a] = s2;
}

// K0b: pack W into WMMA B-fragment layout (f16).
// flat idx = (((r*4+ki)*4+nt)*32+lane)*16+h ; element = W[r][ki*32+(h>=8?16:0)+(lane>=16?8:0)+(h&7)][nt*16+(lane&15)]
__global__ void k_wpack(const float* __restrict__ W, _Float16* __restrict__ wp, int total) {
    int idx = blockIdx.x * 256 + threadIdx.x;
    if (idx >= total) return;
    int h    = idx & 15;
    int lane = (idx >> 4) & 31;
    int nt   = (idx >> 9) & 3;
    int ki   = (idx >> 11) & 3;
    int r    = idx >> 13;
    int k = ki * 32 + ((h >> 3) << 4) + ((lane >> 4) << 3) + (h & 7);
    int n = nt * 16 + (lane & 15);
    wp[idx] = (_Float16)W[((size_t)r * F_IN + k) * F_OUT + n];
}

// K1: relation histogram (block-aggregated). <<<ceil(E/256),256>>>
__global__ void k_hist(const int* __restrict__ erel, int* __restrict__ meta, int E) {
    __shared__ int hc[16];
    int tid = threadIdx.x;
    if (tid < 16) hc[tid] = 0;
    __syncthreads();
    int e = blockIdx.x * 256 + tid;
    if (e < E) atomicAdd(&hc[erel[e]], 1);
    __syncthreads();
    if (tid < RNUM && hc[tid]) atomicAdd(&meta[MI_CNT + tid], hc[tid]);
}

// K5: serial scan over 11 buckets. <<<1,1>>>
__global__ void k_scan(int* __restrict__ meta) {
    int tot = 0, tt = 0;
    for (int r = 0; r < RNUM; ++r) {
        int c = meta[MI_CNT + r];
        meta[MI_BASE + r] = tot;
        meta[MI_CUR + r]  = tot;
        meta[MI_TP + r]   = tt;
        tot += c;
        tt  += (c + 15) >> 4;
    }
    meta[MI_TP + RNUM] = tt;
}

// K2: column max of leakyrelu score, order-invariant encoded atomicMax. <<<NBLK,256>>>
__global__ void k_colmax(const float* __restrict__ x, const float* __restrict__ uv,
                         const int* __restrict__ esrc, const int* __restrict__ edst,
                         const int* __restrict__ erel, int* __restrict__ meta, int E) {
    __shared__ float red[1024];
    int tid = threadIdx.x, cg = tid & 31, rl = tid >> 5;
    const float* u = uv;
    const float* v = uv + RNUM * F_IN;
    float m0 = -3.4e38f, m1 = m0, m2 = m0, m3 = m0;
    for (int e = blockIdx.x * 8 + rl; e < E; e += gridDim.x * 8) {
        int s = esrc[e], d = edst[e], r = erel[e];
        float4 xs = *(const float4*)(x + (size_t)s * F_IN + cg * 4);
        float4 xd = *(const float4*)(x + (size_t)d * F_IN + cg * 4);
        float4 uu = *(const float4*)(u + r * F_IN + cg * 4);
        float4 vv = *(const float4*)(v + r * F_IN + cg * 4);
        m0 = fmaxf(m0, lrelu(xs.x * uu.x + xd.x * vv.x));
        m1 = fmaxf(m1, lrelu(xs.y * uu.y + xd.y * vv.y));
        m2 = fmaxf(m2, lrelu(xs.z * uu.z + xd.z * vv.z));
        m3 = fmaxf(m3, lrelu(xs.w * uu.w + xd.w * vv.w));
    }
    red[tid * 4 + 0] = m0; red[tid * 4 + 1] = m1;
    red[tid * 4 + 2] = m2; red[tid * 4 + 3] = m3;
    __syncthreads();
    if (tid < F_IN) {
        int g = tid >> 2, ci = tid & 3;
        float mx = red[g * 4 + ci];
        for (int rr = 1; rr < 8; ++rr) mx = fmaxf(mx, red[(rr * 32 + g) * 4 + ci]);
        atomicMax((unsigned*)(meta + MI_CMAX) + tid, enc_f32(mx));
    }
}

// K3: fixed-order per-block partial column sums of exp(s - max). <<<NBLK,256>>>
__global__ void k_colsum_part(const float* __restrict__ x, const float* __restrict__ uv,
                              const int* __restrict__ esrc, const int* __restrict__ edst,
                              const int* __restrict__ erel, const int* __restrict__ meta,
                              float* __restrict__ PS, int E) {
    __shared__ float red[1024];
    int tid = threadIdx.x, cg = tid & 31, rl = tid >> 5;
    const float* u = uv;
    const float* v = uv + RNUM * F_IN;
    const unsigned* cme = (const unsigned*)(meta + MI_CMAX);
    float c0 = dec_f32(cme[cg * 4 + 0]), c1 = dec_f32(cme[cg * 4 + 1]);
    float c2 = dec_f32(cme[cg * 4 + 2]), c3 = dec_f32(cme[cg * 4 + 3]);
    float a0 = 0.f, a1 = 0.f, a2 = 0.f, a3 = 0.f;
    for (int e = blockIdx.x * 8 + rl; e < E; e += gridDim.x * 8) {
        int s = esrc[e], d = edst[e], r = erel[e];
        float4 xs = *(const float4*)(x + (size_t)s * F_IN + cg * 4);
        float4 xd = *(const float4*)(x + (size_t)d * F_IN + cg * 4);
        float4 uu = *(const float4*)(u + r * F_IN + cg * 4);
        float4 vv = *(const float4*)(v + r * F_IN + cg * 4);
        a0 += __expf(lrelu(xs.x * uu.x + xd.x * vv.x) - c0);
        a1 += __expf(lrelu(xs.y * uu.y + xd.y * vv.y) - c1);
        a2 += __expf(lrelu(xs.z * uu.z + xd.z * vv.z) - c2);
        a3 += __expf(lrelu(xs.w * uu.w + xd.w * vv.w) - c3);
    }
    red[tid * 4 + 0] = a0; red[tid * 4 + 1] = a1;
    red[tid * 4 + 2] = a2; red[tid * 4 + 3] = a3;
    __syncthreads();
    if (tid < F_IN) {
        int g = tid >> 2, ci = tid & 3;
        float sm = 0.f;
        for (int rr = 0; rr < 8; ++rr) sm += red[(rr * 32 + g) * 4 + ci];  // fixed order
        PS[(size_t)blockIdx.x * F_IN + tid] = sm;
    }
}

// K3b: deterministic tree reduction of NBLK partials per column. <<<128,256>>>
__global__ void k_colsum_red(const float* __restrict__ PS, int* __restrict__ meta) {
    __shared__ float red[256];
    int c = blockIdx.x, t = threadIdx.x;
    float s = 0.f;
    for (int j = 0; j < NBLK / 256; ++j) s += PS[(size_t)(j * 256 + t) * F_IN + c];
    red[t] = s;
    __syncthreads();
    for (int st = 128; st > 0; st >>= 1) {
        if (t < st) red[t] += red[t + st];
        __syncthreads();
    }
    if (t == 0) ((float*)(meta + MI_CSUM))[c] = red[0];
}

// K4: z = softmax * x_dst, scaled, stored f16 row-major [E][128]. <<<ceil(E*32/256),256>>>
__global__ void k_zwrite(const float* __restrict__ x, const float* __restrict__ uv,
                         const int* __restrict__ esrc, const int* __restrict__ edst,
                         const int* __restrict__ erel, const int* __restrict__ meta,
                         _Float16* __restrict__ zH, int E) {
    long idx = (long)blockIdx.x * 256 + threadIdx.x;
    if (idx >= (long)E * 32) return;
    int e = (int)(idx >> 5), cg = (int)(idx & 31);
    const float* u = uv;
    const float* v = uv + RNUM * F_IN;
    const unsigned* cme = (const unsigned*)(meta + MI_CMAX);
    const float* csum = (const float*)(meta + MI_CSUM);
    int s = esrc[e], d = edst[e], r = erel[e];
    float4 xs = *(const float4*)(x + (size_t)s * F_IN + cg * 4);
    float4 xd = *(const float4*)(x + (size_t)d * F_IN + cg * 4);
    float4 uu = *(const float4*)(u + r * F_IN + cg * 4);
    float4 vv = *(const float4*)(v + r * F_IN + cg * 4);
    float z0 = __expf(lrelu(xs.x * uu.x + xd.x * vv.x) - dec_f32(cme[cg * 4 + 0])) * (ZSCALE / csum[cg * 4 + 0]) * xd.x;
    float z1 = __expf(lrelu(xs.y * uu.y + xd.y * vv.y) - dec_f32(cme[cg * 4 + 1])) * (ZSCALE / csum[cg * 4 + 1]) * xd.y;
    float z2 = __expf(lrelu(xs.z * uu.z + xd.z * vv.z) - dec_f32(cme[cg * 4 + 2])) * (ZSCALE / csum[cg * 4 + 2]) * xd.z;
    float z3 = __expf(lrelu(xs.w * uu.w + xd.w * vv.w) - dec_f32(cme[cg * 4 + 3])) * (ZSCALE / csum[cg * 4 + 3]) * xd.w;
    h4 o;
    o[0] = (_Float16)z0; o[1] = (_Float16)z1; o[2] = (_Float16)z2; o[3] = (_Float16)z3;
    *(h4*)(zH + (size_t)e * F_IN + cg * 4) = o;
}

// K6: block-aggregated counting-sort scatter. <<<ceil(E/256),256>>>
__global__ void k_scatter(const int* __restrict__ erel, int* __restrict__ meta,
                          int* __restrict__ sorted, int E) {
    __shared__ int hc[16];
    __shared__ int hbase[16];
    int tid = threadIdx.x;
    if (tid < 16) hc[tid] = 0;
    __syncthreads();
    int e = blockIdx.x * 256 + tid, r = 0, rank = 0;
    if (e < E) { r = erel[e]; rank = atomicAdd(&hc[r], 1); }
    __syncthreads();
    if (tid < RNUM) {
        int c = hc[tid];
        hbase[tid] = c ? atomicAdd(&meta[MI_CUR + tid], c) : 0;
    }
    __syncthreads();
    if (e < E) sorted[hbase[r] + rank] = e;
}

// K7: WMMA GEMM per 16-edge tile: o[e,:] = z[e,:] @ W[rel[e]]  (M16 x N64, K=128)
// one wave per block; 16x v_wmma_f32_16x16x32_f16 per block. <<<Tmax,32>>>
__global__ void k_gemm(const int* __restrict__ meta, const int* __restrict__ sorted,
                       const _Float16* __restrict__ zH, const _Float16* __restrict__ wp,
                       float* __restrict__ out) {
    int t = blockIdx.x;
    const int* tp = meta + MI_TP;
    if (t >= tp[RNUM]) return;
    int r = 0;
    while (r + 1 < RNUM && tp[r + 1] <= t) r++;
    int lt = t - tp[r];
    int base = meta[MI_BASE + r];
    int rowStart = base + lt * 16;
    int segEnd = base + meta[MI_CNT + r];
    int lane = threadIdx.x;
    int m = lane & 15;
    int hi = lane >> 4;
    int row = rowStart + m;
    bool rv = row < segEnd;
    int edge = rv ? sorted[row] : 0;
    const _Float16* zp = zH + (size_t)edge * F_IN + (hi << 3);
    v8f c[4] = {};
#pragma unroll
    for (int ki = 0; ki < 4; ++ki) {
        h16 a = {};
        if (rv) {
            h8 lo = *(const h8*)(zp + ki * 32);
            h8 hh = *(const h8*)(zp + ki * 32 + 16);
            a = __builtin_shufflevector(lo, hh, 0, 1, 2, 3, 4, 5, 6, 7, 8, 9, 10, 11, 12, 13, 14, 15);
        }
#pragma unroll
        for (int nt = 0; nt < 4; ++nt) {
            const _Float16* bp = wp + ((size_t)(((r * 4 + ki) * 4 + nt) * 32 + lane) << 4);
            h8 blo = *(const h8*)bp;
            h8 bhi = *(const h8*)(bp + 8);
            h16 b = __builtin_shufflevector(blo, bhi, 0, 1, 2, 3, 4, 5, 6, 7, 8, 9, 10, 11, 12, 13, 14, 15);
            c[nt] = __builtin_amdgcn_wmma_f32_16x16x32_f16(false, a, false, b, (short)0, c[nt], false, false);
        }
    }
    int n = lane & 15;
#pragma unroll
    for (int j = 0; j < 8; ++j) {
        int rw = rowStart + hi * 8 + j;
        if (rw < segEnd) {
            int e2 = sorted[rw];
            float* op = out + (size_t)e2 * F_OUT + n;
#pragma unroll
            for (int nt = 0; nt < 4; ++nt) {
                float val = c[nt][j] * INV_ZSCALE;
                op[nt * 16] = val > 0.0f ? val : expm1f(val);   // ELU
            }
        }
    }
}

extern "C" void kernel_launch(void* const* d_in, const int* in_sizes, int n_in,
                              void* d_out, int out_size, void* d_ws, size_t ws_size,
                              hipStream_t stream) {
    const float* x    = (const float*)d_in[0];
    const float* W    = (const float*)d_in[1];
    const float* aw   = (const float*)d_in[2];
    const int*   esrc = (const int*)d_in[3];
    const int*   edst = (const int*)d_in[4];
    const int*   erel = (const int*)d_in[5];
    float* out = (float*)d_out;
    const int E = in_sizes[3];

    char* ws = (char*)d_ws;
    int*      meta   = (int*)(ws + META_OFF);
    float*    uv     = (float*)(ws + UV_OFF);
    _Float16* wp     = (_Float16*)(ws + WP_OFF);
    float*    PS     = (float*)(ws + PS_OFF);
    int*      sorted = (int*)(ws + SORT_OFF);
    _Float16* zH     = (_Float16*)(ws + ZH_OFF);

    hipMemsetAsync(meta, 0, 2048, stream);

    k_uv<<<RNUM, 128, 0, stream>>>(W, aw, uv);
    const int wpTotal = RNUM * 4 * 4 * 32 * 16;
    k_wpack<<<(wpTotal + 255) / 256, 256, 0, stream>>>(W, wp, wpTotal);

    const int eb = (E + 255) / 256;
    k_hist<<<eb, 256, 0, stream>>>(erel, meta, E);
    k_scan<<<1, 1, 0, stream>>>(meta);

    k_colmax<<<NBLK, 256, 0, stream>>>(x, uv, esrc, edst, erel, meta, E);
    k_colsum_part<<<NBLK, 256, 0, stream>>>(x, uv, esrc, edst, erel, meta, PS, E);
    k_colsum_red<<<F_IN, 256, 0, stream>>>(PS, meta);

    long zThreads = (long)E * 32;
    k_zwrite<<<(int)((zThreads + 255) / 256), 256, 0, stream>>>(x, uv, esrc, edst, erel, meta, zH, E);

    k_scatter<<<eb, 256, 0, stream>>>(erel, meta, sorted, E);

    const int Tmax = (E + 15) / 16 + RNUM;   // upper bound; excess blocks exit on tp[RNUM]
    k_gemm<<<Tmax, 32, 0, stream>>>(meta, sorted, zH, wp, out);
}
